// GFN_2791728742503
// MI455X (gfx1250) — matrix operand order repaired and checked by reading, not compiled, not run on hardware
//
#include <hip/hip_runtime.h>
#include <hip/hip_bf16.h>
#include <math.h>

// ---------------------------------------------------------------------------
// GFN diffusion sampler for MI455X (gfx1250, wave32, WMMA + async LDS DMA).
// Trajectories independent -> parallelize over batch, loop T inside one
// persistent kernel. Weights fp16-transposed in LDS, state f32 in LDS,
// matmuls via v_wmma_f32_16x16x32_f16, noise prefetched and states streamed
// with global_load_async_to_lds_b128 / global_store_async_from_lds_b128.
// ---------------------------------------------------------------------------

typedef __attribute__((ext_vector_type(16))) _Float16 v16h;
typedef __attribute__((ext_vector_type(8)))  _Float16 v8h;
typedef __attribute__((ext_vector_type(4)))  _Float16 v4h;
typedef __attribute__((ext_vector_type(2)))  _Float16 v2h;
typedef __attribute__((ext_vector_type(8)))  float    v8f;

#define TCNT     100
#define BSZX     4096
#define DTF      0.01f
#define SQRT_DT  0.1f
#define LOG_DT   (-4.6051701859880914f)
#define LOGTWOPI 1.8378770664093453f

// LDS weight layout (fp16, transposed: [N][K]), offsets in halfs
#define OFF_SE1 0       // 64x128
#define OFF_SE2 8192    // 64x64
#define OFF_J1  12288   // 64x128
#define OFF_J2  20480   // 64x64
#define OFF_J3  24576   // 256x64
#define OFF_F1  40960   // 64x128
#define OFF_F2  49152   // 64x64
#define OFF_F3  53248   // 16x64 (row 0 real, rest zero)
#define W_TOTAL 54272

// bias offsets (f32)
#define BO_SE1 0
#define BO_SE2 64
#define BO_J1  128
#define BO_J2  192
#define BO_J3  256   // 256 entries
#define BO_F1  512
#define BO_F2  576
#define BO_F3  640

// ---- branchless fast math on HW transcendental units ----------------------
__device__ __forceinline__ float fast_exp(float x) {
    return __builtin_amdgcn_exp2f(x * 1.4426950408889634f);
}
__device__ __forceinline__ float fast_log(float x) {
    return __builtin_amdgcn_logf(x) * 0.6931471805599453f;
}
// Abramowitz-Stegun 7.1.26, |err| <= 1.5e-7, fully branchless
__device__ __forceinline__ float erf_fast(float x) {
    float ax = fabsf(x);
    float t  = __builtin_amdgcn_rcpf(fmaf(0.3275911f, ax, 1.0f));
    float p  = fmaf(1.061405429f, t, -1.453152027f);
    p = fmaf(p, t, 1.421413741f);
    p = fmaf(p, t, -0.284496736f);
    p = fmaf(p, t, 0.254829592f);
    p = p * t;
    float e = fast_exp(-ax * ax);
    float r = fmaf(-p, e, 1.0f);
    return copysignf(r, x);
}
__device__ __forceinline__ float gelu_fast(float x) {
    return 0.5f * x * (1.0f + erf_fast(x * 0.70710678118654752f));
}
__device__ __forceinline__ float tanh_fast(float x) {
    float e = fast_exp(-2.0f * fabsf(x));
    float r = (1.0f - e) * __builtin_amdgcn_rcpf(1.0f + e);
    return copysignf(r, x);
}

// ---- CDNA5 async global<->LDS DMA (ASYNCcnt) ------------------------------
__device__ __forceinline__ void async_load_b128(unsigned lds_addr, const void* base,
                                                unsigned voff) {
    asm volatile("global_load_async_to_lds_b128 %0, %1, %2"
                 :: "v"(lds_addr), "v"(voff), "s"(base) : "memory");
}
__device__ __forceinline__ void async_store_b128(unsigned lds_addr, void* base,
                                                 unsigned voff) {
    asm volatile("global_store_async_from_lds_b128 %0, %1, %2"
                 :: "v"(voff), "v"(lds_addr), "s"(base) : "memory");
}
__device__ __forceinline__ void wait_async0() {
    asm volatile("s_wait_asynccnt 0x0" ::: "memory");
}
__device__ __forceinline__ void wait_ds0() {
    asm volatile("s_wait_dscnt 0x0" ::: "memory");
}
__device__ __forceinline__ unsigned lds_addr_of(const void* p) {
    return (unsigned)(size_t)p; // generic->LDS: low 32 bits are LDS byte offset
}

// ---- WMMA fragment loaders (ISA 7.12.2 layouts) ---------------------------
__device__ __forceinline__ v16h load_a(const _Float16* base, int stride, int k0, int lane) {
    const _Float16* p = base + (lane & 15) * stride + k0 + ((lane >> 4) << 3);
    v8h lo = *(const v8h*)(p);
    v8h hi = *(const v8h*)(p + 16);
    return __builtin_shufflevector(lo, hi, 0,1,2,3,4,5,6,7,8,9,10,11,12,13,14,15);
}
__device__ __forceinline__ v16h load_b(const _Float16* Wt, int K, int n0, int k0, int lane) {
    const _Float16* p = Wt + (n0 + (lane & 15)) * K + k0 + ((lane >> 4) << 3);
    v8h lo = *(const v8h*)(p);
    v8h hi = *(const v8h*)(p + 16);
    return __builtin_shufflevector(lo, hi, 0,1,2,3,4,5,6,7,8,9,10,11,12,13,14,15);
}
__device__ __forceinline__ v8f wmma_f16(v16h a, v16h b, v8f c) {
    return __builtin_amdgcn_wmma_f32_16x16x32_f16(false, a, false, b, (short)0, c, false, false);
}

// One dense layer: out[16][N] = gelu(A[16][K] @ Wt^T + bias), fp16 to dst.
template<int K, int N>
__device__ __forceinline__ void mlp_layer(const _Float16* Aact, int As,
                                          const _Float16* Wt, const float* bias,
                                          _Float16* dst, int Ds, int lane) {
    const int nidx = lane & 15;
    const int roff = (lane >> 4) << 3;
    v16h a[K / 32];
    #pragma unroll
    for (int k = 0; k < K / 32; ++k) a[k] = load_a(Aact, As, k * 32, lane);
    #pragma unroll
    for (int n0 = 0; n0 < N; n0 += 16) {
        v8f acc = {};
        #pragma unroll
        for (int k = 0; k < K / 32; ++k)
            acc = wmma_f16(a[k], load_b(Wt, K, n0, k * 32, lane), acc);
        int n = n0 + nidx;
        float bi = bias[n];
        #pragma unroll
        for (int v = 0; v < 8; ++v)
            dst[(v + roff) * Ds + n] = (_Float16)gelu_fast(acc[v] + bi);
    }
}

template<int K, int N>
__device__ __forceinline__ void load_wt_tr(_Float16* dst, const float* g, int tid, int nth) {
    for (int idx = tid; idx < K * N; idx += nth) {
        int n = idx / K;
        int k = idx - n * K;
        dst[idx] = (_Float16)g[k * N + n];
    }
}

// ---------------------------------------------------------------------------
// Kernel 1: t_emb[T][64] fp16
// ---------------------------------------------------------------------------
__global__ __launch_bounds__(64) void gfn_temb_kernel(
    const float* __restrict__ tW1, const float* __restrict__ tb1,
    const float* __restrict__ tW2, const float* __restrict__ tb2,
    _Float16* __restrict__ temb) {
    __shared__ float h1[64];
    const int t = blockIdx.x;
    const int j = threadIdx.x;
    const float ts = (float)t * DTF;
    float acc = tb1[j];
    for (int k = 0; k < 128; ++k) {
        float ang = ts * (float)((k & 63) + 1) * 6.28318530717958647692f;
        float te = (k < 64) ? sinf(ang) : cosf(ang);
        acc += te * tW1[k * 64 + j];
    }
    h1[j] = gelu_fast(acc);
    __syncthreads();
    float a2 = tb2[j];
    for (int k = 0; k < 64; ++k) a2 += h1[k] * tW2[k * 64 + j];
    temb[t * 64 + j] = (_Float16)gelu_fast(a2);
}

// ---------------------------------------------------------------------------
// Kernel 2: persistent scan. 64 threads = 2 waves/block; each wave owns a
// 16-row batch tile for all 100 steps.
// ---------------------------------------------------------------------------
__global__ __launch_bounds__(64) void gfn_main_kernel(
    const float* __restrict__ s_in, const float* __restrict__ noise,
    const float* __restrict__ seW1, const float* __restrict__ seb1,
    const float* __restrict__ seW2, const float* __restrict__ seb2,
    const float* __restrict__ jW1,  const float* __restrict__ jb1,
    const float* __restrict__ jW2,  const float* __restrict__ jb2,
    const float* __restrict__ jW3,  const float* __restrict__ jb3,
    const float* __restrict__ fW1,  const float* __restrict__ fb1,
    const float* __restrict__ fW2,  const float* __restrict__ fb2,
    const float* __restrict__ fW3,  const float* __restrict__ fb3,
    const _Float16* __restrict__ temb,
    float* __restrict__ states, float* __restrict__ logpf,
    float* __restrict__ logpb, float* __restrict__ logfl) {

    __shared__ __align__(16) _Float16 sW[W_TOTAL];
    __shared__ float sB[644];
    __shared__ __align__(16) float    Sst[2][16 * 128];
    __shared__ __align__(16) float    Nb[2][16 * 128];   // async noise staging
    __shared__ __align__(16) _Float16 Ah[2][16 * 128];
    __shared__ __align__(16) _Float16 Cb[2][16 * 64];
    __shared__ __align__(16) _Float16 Db[2][16 * 64];

    const int tid = threadIdx.x;
    const int lane = tid & 31;
    const int w = tid >> 5;
    const int nidx = lane & 15;
    const int roff = (lane >> 4) << 3;

    // cooperative weight (transpose->fp16) + bias load
    load_wt_tr<128, 64>(sW + OFF_SE1, seW1, tid, 64);
    load_wt_tr<64, 64>(sW + OFF_SE2, seW2, tid, 64);
    load_wt_tr<128, 64>(sW + OFF_J1, jW1, tid, 64);
    load_wt_tr<64, 64>(sW + OFF_J2, jW2, tid, 64);
    load_wt_tr<64, 256>(sW + OFF_J3, jW3, tid, 64);
    load_wt_tr<128, 64>(sW + OFF_F1, fW1, tid, 64);
    load_wt_tr<64, 64>(sW + OFF_F2, fW2, tid, 64);
    for (int idx = tid; idx < 16 * 64; idx += 64) {
        int n = idx >> 6, k = idx & 63;
        sW[OFF_F3 + idx] = (n == 0) ? (_Float16)fW3[k] : (_Float16)0.0f;
    }
    sB[BO_SE1 + tid] = seb1[tid];
    sB[BO_SE2 + tid] = seb2[tid];
    sB[BO_J1 + tid] = jb1[tid];
    sB[BO_J2 + tid] = jb2[tid];
    #pragma unroll
    for (int q = 0; q < 4; ++q) sB[BO_J3 + q * 64 + tid] = jb3[q * 64 + tid];
    sB[BO_F1 + tid] = fb1[tid];
    sB[BO_F2 + tid] = fb2[tid];
    if (tid == 0) sB[BO_F3] = fb3[0];
    __syncthreads();

    float* Sw = Sst[w];
    float* Nw = Nb[w];
    _Float16* Aw = Ah[w];
    _Float16* Cw = Cb[w];
    _Float16* Dw = Db[w];
    const unsigned SwLds = lds_addr_of(Sw);
    const unsigned NwLds = lds_addr_of(Nw);

    const int tile = blockIdx.x * 2 + w;
    const int b0 = tile * 16;

    // init state + states[:,0,:]  (b128 vectorized)
    for (int idx = lane; idx < 512; idx += 32) {
        int r = idx >> 5, c = (idx & 31) << 2;
        float4 v = *(const float4*)(s_in + (size_t)(b0 + r) * 128 + c);
        ((float4*)Sw)[idx] = v;
        *(float4*)(states + (size_t)(b0 + r) * (101 * 128) + c) = v;
    }

    for (int t = 0; t < TCNT; ++t) {
        // async prefetch of this step's noise tile (HBM -> LDS, ASYNCcnt),
        // overlapped with the whole MLP chain below
        {
            unsigned goff = (unsigned)t * (BSZX * 128u * 4u) + (unsigned)b0 * 512u;
            #pragma unroll
            for (int i = 0; i < 16; ++i) {
                unsigned idx = (unsigned)lane + 32u * i;
                async_load_b128(NwLds + idx * 16u, (const void*)noise, goff + idx * 16u);
            }
        }

        // s_cur -> fp16 A (packed b64 stores)
        for (int idx = lane; idx < 512; idx += 32) {
            float4 f = ((const float4*)Sw)[idx];
            v4h h = { (_Float16)f.x, (_Float16)f.y, (_Float16)f.z, (_Float16)f.w };
            ((v4h*)Aw)[idx] = h;
        }

        // state-embed MLP
        mlp_layer<128, 64>(Aw, 128, sW + OFF_SE1, sB + BO_SE1, Cw, 64, lane);
        mlp_layer<64, 64>(Cw, 64, sW + OFF_SE2, sB + BO_SE2, Aw, 128, lane); // cols 0..63

        // st = [s_emb | t_e] : broadcast t_e into cols 64..127 (b32 stores)
        {
            v2h tv = ((const v2h*)(temb + t * 64))[lane];
            #pragma unroll
            for (int r = 0; r < 16; ++r)
                ((v2h*)(Aw + r * 128 + 64))[lane] = tv;
        }

        // flow branch
        mlp_layer<128, 64>(Aw, 128, sW + OFF_F1, sB + BO_F1, Cw, 64, lane);
        mlp_layer<64, 64>(Cw, 64, sW + OFF_F2, sB + BO_F2, Dw, 64, lane);
        {
            v16h a0 = load_a(Dw, 64, 0, lane);
            v16h a1 = load_a(Dw, 64, 32, lane);
            v8f accF = {};
            accF = wmma_f16(a0, load_b(sW + OFF_F3, 64, 0, 0, lane), accF);
            accF = wmma_f16(a1, load_b(sW + OFF_F3, 64, 0, 32, lane), accF);
            if (nidx == 0) {
                float bf = sB[BO_F3];
                #pragma unroll
                for (int v = 0; v < 8; ++v)
                    logfl[(size_t)(b0 + v + roff) * 101 + t] = accF[v] + bf;
            }
        }

        // joint branch
        mlp_layer<128, 64>(Aw, 128, sW + OFF_J1, sB + BO_J1, Cw, 64, lane);
        mlp_layer<64, 64>(Cw, 64, sW + OFF_J2, sB + BO_J2, Dw, 64, lane);

        // wait: noise tile arrived AND previous step's state stores drained
        // (Sw is overwritten below)
        wait_async0();

        // j3 (16x256) in 16-col chunks of (mean, logvar) + elementwise update
        float spf[8], spb[8];
        #pragma unroll
        for (int v = 0; v < 8; ++v) { spf[v] = 0.0f; spb[v] = 0.0f; }
        const float ratio = (float)t / (float)(t + 1);
        const float rinvb = (t > 0) ? __builtin_amdgcn_rsqf(DTF * ratio) : 0.0f;

        v16h aj0 = load_a(Dw, 64, 0, lane);
        v16h aj1 = load_a(Dw, 64, 32, lane);

        for (int nd = 0; nd < 8; ++nd) {
            v8f accM = {};
            v8f accV = {};
            accM = wmma_f16(aj0, load_b(sW + OFF_J3, 64, nd * 16, 0, lane), accM);
            accM = wmma_f16(aj1, load_b(sW + OFF_J3, 64, nd * 16, 32, lane), accM);
            accV = wmma_f16(aj0, load_b(sW + OFF_J3, 64, 128 + nd * 16, 0, lane), accV);
            accV = wmma_f16(aj1, load_b(sW + OFF_J3, 64, 128 + nd * 16, 32, lane), accV);

            const int d = nd * 16 + nidx;
            const float bM = sB[BO_J3 + d];
            const float bV = sB[BO_J3 + 128 + d];
            #pragma unroll
            for (int v = 0; v < 8; ++v) {
                int r = v + roff;
                float mean = accM[v] + bM;
                float lv = tanh_fast(accV[v] + bV) * 4.0f;
                float sc = Sw[r * 128 + d];
                float ep = Nw[r * 128 + d];
                float sig = SQRT_DT * fast_exp(0.5f * lv);
                float sn = sc + DTF * mean + sig * ep;
                float z = (sn - sc - DTF * mean) * __builtin_amdgcn_rcpf(sig);
                spf[v] += z * z + lv;
                float zb = (sc - sn * ratio) * rinvb;
                spb[v] += zb * zb; // rinvb==0 at t==0
                Sw[r * 128 + d] = sn;
            }
        }

        // reduce per-row sums across the 16-lane N groups (wave32 safe)
        #pragma unroll
        for (int m = 1; m < 16; m <<= 1) {
            #pragma unroll
            for (int v = 0; v < 8; ++v) {
                spf[v] += __shfl_xor(spf[v], m, 32);
                spb[v] += __shfl_xor(spb[v], m, 32);
            }
        }
        if (nidx == 0) {
            float cpb = (t > 0) ? 128.0f * (LOGTWOPI + fast_log(DTF * ratio)) : 0.0f;
            #pragma unroll
            for (int v = 0; v < 8; ++v) {
                int br = b0 + v + roff;
                logpf[(size_t)br * 100 + t] =
                    -0.5f * (spf[v] + 128.0f * (LOGTWOPI + LOG_DT));
                logpb[(size_t)br * 100 + t] =
                    (t > 0) ? -0.5f * (spb[v] + cpb) : 0.0f;
            }
        }

        // stream states[:, t+1, :] straight from LDS (LDS -> HBM, ASYNCcnt),
        // overlapped with the next step's MLP chain. DS stores must land first.
        wait_ds0();
        {
            #pragma unroll
            for (int i = 0; i < 16; ++i) {
                unsigned idx = (unsigned)lane + 32u * i;
                unsigned r = idx >> 5, c = (idx & 31) << 2;
                unsigned goff = ((unsigned)b0 + r) * (101u * 128u * 4u) +
                                (unsigned)(t + 1) * 512u + c * 4u;
                async_store_b128(SwLds + idx * 16u, (void*)states, goff);
            }
        }
    }

    // logf last column = 0  (final async stores drained by S_ENDPGM wait-idle)
    if (lane < 16)
        logfl[(size_t)(b0 + lane) * 101 + 100] = 0.0f;
}

extern "C" void kernel_launch(void* const* d_in, const int* in_sizes, int n_in,
                              void* d_out, int out_size, void* d_ws, size_t ws_size,
                              hipStream_t stream) {
    (void)in_sizes; (void)n_in; (void)out_size; (void)ws_size;
    const float* s     = (const float*)d_in[0];
    const float* noise = (const float*)d_in[1];
    const float* tW1 = (const float*)d_in[2];
    const float* tb1 = (const float*)d_in[3];
    const float* tW2 = (const float*)d_in[4];
    const float* tb2 = (const float*)d_in[5];
    const float* seW1 = (const float*)d_in[6];
    const float* seb1 = (const float*)d_in[7];
    const float* seW2 = (const float*)d_in[8];
    const float* seb2 = (const float*)d_in[9];
    const float* jW1 = (const float*)d_in[10];
    const float* jb1 = (const float*)d_in[11];
    const float* jW2 = (const float*)d_in[12];
    const float* jb2 = (const float*)d_in[13];
    const float* jW3 = (const float*)d_in[14];
    const float* jb3 = (const float*)d_in[15];
    const float* fW1 = (const float*)d_in[16];
    const float* fb1 = (const float*)d_in[17];
    const float* fW2 = (const float*)d_in[18];
    const float* fb2 = (const float*)d_in[19];
    const float* fW3 = (const float*)d_in[20];
    const float* fb3 = (const float*)d_in[21];

    float* out = (float*)d_out;
    float* states = out;                              // 4096*101*128
    float* logpf  = out + (size_t)4096 * 101 * 128;   // 4096*100
    float* logpb  = logpf + (size_t)4096 * 100;       // 4096*100
    float* logfl  = logpb + (size_t)4096 * 100;       // 4096*101

    _Float16* temb = (_Float16*)d_ws;                 // 100*64 fp16

    gfn_temb_kernel<<<dim3(100), dim3(64), 0, stream>>>(tW1, tb1, tW2, tb2, temb);
    gfn_main_kernel<<<dim3(128), dim3(64), 0, stream>>>(
        s, noise, seW1, seb1, seW2, seb2,
        jW1, jb1, jW2, jb2, jW3, jb3,
        fW1, fb1, fW2, fb2, fW3, fb3,
        temb, states, logpf, logpb, logfl);
}
